// Attention_738734375692
// MI455X (gfx1250) — matrix operand order repaired and checked by reading, not compiled, not run on hardware
//
#include <hip/hip_runtime.h>
#include <hip/hip_bf16.h>

// ---------------------------------------------------------------------------
// Multi-head attention (B=8, N=1024, C=768, h=12, d=64) for gfx1250 (MI455X).
// All four matmuls run through v_wmma_f32_16x16x32_bf16 (wave32 WMMA).
// GEMMs use 2x2 register tiling: 32x32 output per wave, 4 independent
// accumulator chains, 2x operand reuse per fragment.
// ---------------------------------------------------------------------------

typedef __bf16 bf16;
typedef bf16  bf16x4 __attribute__((ext_vector_type(4)));
typedef bf16  bf16x8 __attribute__((ext_vector_type(8)));
typedef bf16  v16bf  __attribute__((ext_vector_type(16)));
typedef float v8f    __attribute__((ext_vector_type(8)));

#define NHEAD 12
#define DHEAD 64
#define CDIM  768
#define NSEQ  1024
#define BATCH 8
#define ROWS  (BATCH * NSEQ)   /* 8192 */
#define OC3   (3 * CDIM)       /* 2304 */

static __device__ __forceinline__ v8f wmma_bf16(v16bf a, v16bf b, v8f c) {
  // D = A(16x32 bf16) * B(32x16 bf16) + C(16x16 f32)
  return __builtin_amdgcn_wmma_f32_16x16x32_bf16(
      /*neg_a=*/false, a, /*neg_b=*/false, b,
      /*c_mod=*/(short)0, c, /*reuse_a=*/false, /*reuse_b=*/false);
}

// A-fragment (16x32, 16-bit): lane l holds row M=l&15; half=l>>4 selects the
// K sub-run. Elements 0..7 = K = kb..kb+7, elements 8..15 = K = kb+8..kb+15
// with kb = half*16. Two contiguous 16-byte loads per lane.
static __device__ __forceinline__ v16bf load_a_frag(const bf16* lo, const bf16* hi) {
  v16bf v;
  *(bf16x8*)&v         = *(const bf16x8*)lo;
  *(((bf16x8*)&v) + 1) = *(const bf16x8*)hi;
  return v;
}

// ---------------------------------------------------------------------------
// Kernel 0: f32 -> bf16 conversion (vectorized x4; all sizes divisible by 4)
// ---------------------------------------------------------------------------
__global__ void cvt_f32_bf16(const float* __restrict__ src,
                             bf16* __restrict__ dst, int n) {
  int i = (blockIdx.x * blockDim.x + threadIdx.x) * 4;
  if (i + 3 < n) {
    float4 v = *(const float4*)(src + i);
    bf16x4 o = { (bf16)v.x, (bf16)v.y, (bf16)v.z, (bf16)v.w };
    *(bf16x4*)(dst + i) = o;
  }
}

// ---------------------------------------------------------------------------
// QKV epilogue for one 16x16 C/D tile: lane l, vgpr rr holds element
// (M = rr + 8*half, N = l&15).  `col` already includes the lane column.
// ---------------------------------------------------------------------------
static __device__ __forceinline__ void store_qkv_tile(
    const v8f& acc, int row0, int col, int half,
    const float* __restrict__ qkv_b,
    bf16* __restrict__ Q, bf16* __restrict__ K, bf16* __restrict__ Vt) {
  const float bias  = qkv_b[col];
  const int   which = col / CDIM;      // 0=Q 1=K 2=V (32-wide tiles never straddle)
  const int   rem   = col % CDIM;
  const int   hh    = rem >> 6;
  const int   dd    = rem & 63;
  const float qscale = 0.125f;         // d^-0.5, d = 64
#pragma unroll
  for (int rr = 0; rr < 8; ++rr) {
    const int row = row0 + rr + half * 8;   // global token index
    const int b_  = row >> 10;
    const int n   = row & 1023;
    const float v = acc[rr] + bias;
    if (which == 0) {
      Q[(((size_t)b_ * NHEAD + hh) * NSEQ + n) * DHEAD + dd] = (bf16)(v * qscale);
    } else if (which == 1) {
      K[(((size_t)b_ * NHEAD + hh) * NSEQ + n) * DHEAD + dd] = (bf16)v;
    } else {
      Vt[(((size_t)b_ * NHEAD + hh) * DHEAD + dd) * NSEQ + n] = (bf16)v;
    }
  }
}

// ---------------------------------------------------------------------------
// Kernel 1: fused QKV GEMM.  [8192 x 768] (bf16) x [2304 x 768]^T (bf16) + b.
// One wave per 32x32 output block (2x2 WMMA tiles, 4 accumulator chains).
// Q is pre-scaled by d^-0.5, V is written transposed ([b,h,d,n]) so attention
// B-fragments are contiguous loads.
// ---------------------------------------------------------------------------
__global__ void __launch_bounds__(256)
qkv_gemm(const bf16* __restrict__ xb, const bf16* __restrict__ wqkv,
         const float* __restrict__ qkv_b,
         bf16* __restrict__ Q, bf16* __restrict__ K, bf16* __restrict__ Vt) {
  const int wave  = blockIdx.x * 8 + (threadIdx.x >> 5);
  const int lane  = threadIdx.x & 31;
  const int r     = lane & 15;
  const int half  = lane >> 4;
  const int tileC = wave % (OC3 / 32);           // 0..71
  const int tileR = wave / (OC3 / 32);           // 0..255
  const int row0  = tileR * 32;
  const int o0    = tileC * 32;

  const bf16* arow0 = xb + (size_t)(row0 + r) * CDIM;
  const bf16* arow1 = arow0 + (size_t)16 * CDIM;
  const bf16* brow0 = wqkv + (size_t)(o0 + r) * CDIM;   // B col = weight row
  const bf16* brow1 = brow0 + (size_t)16 * CDIM;

  v8f acc00 = {}, acc01 = {}, acc10 = {}, acc11 = {};
#pragma unroll 2
  for (int c0 = 0; c0 < CDIM; c0 += 32) {
    v16bf a0 = load_a_frag(arow0 + c0 + half * 16, arow0 + c0 + half * 16 + 8);
    v16bf a1 = load_a_frag(arow1 + c0 + half * 16, arow1 + c0 + half * 16 + 8);
    v16bf b0 = *(const v16bf*)(brow0 + c0 + half * 16);
    v16bf b1 = *(const v16bf*)(brow1 + c0 + half * 16);
    acc00 = wmma_bf16(a0, b0, acc00);
    acc01 = wmma_bf16(a0, b1, acc01);
    acc10 = wmma_bf16(a1, b0, acc10);
    acc11 = wmma_bf16(a1, b1, acc11);
  }

  store_qkv_tile(acc00, row0,      o0 + r,      half, qkv_b, Q, K, Vt);
  store_qkv_tile(acc01, row0,      o0 + 16 + r, half, qkv_b, Q, K, Vt);
  store_qkv_tile(acc10, row0 + 16, o0 + r,      half, qkv_b, Q, K, Vt);
  store_qkv_tile(acc11, row0 + 16, o0 + 16 + r, half, qkv_b, Q, K, Vt);
}

// ---------------------------------------------------------------------------
// Kernel 2: flash attention.  One wave owns 16 queries of one (b,h); streams
// keys 32 at a time: 4 WMMAs for S = Q K^T, online softmax through a per-wave
// LDS slice, 4 WMMAs for O += P V.  Result -> attn buffer [8192 x 768] bf16.
// ---------------------------------------------------------------------------
__global__ void __launch_bounds__(256)
flash_attn(const bf16* __restrict__ Q, const bf16* __restrict__ K,
           const bf16* __restrict__ Vt, bf16* __restrict__ attn) {
  __shared__ float sS[8][16][32];   // raw scores
  __shared__ bf16  sP[8][16][32];   // exp(scores) in A-fragment-friendly order
  __shared__ float sAl[8][16];      // per-row rescale factor / final 1/l

  const int widx = threadIdx.x >> 5;
  const int lane = threadIdx.x & 31;
  const int r    = lane & 15;
  const int half = lane >> 4;

  const int w     = blockIdx.x * 8 + widx;      // 0..6143
  const int qtile = w & 63;
  const int bh    = w >> 6;
  const int hh    = bh % NHEAD;
  const int b_    = bh / NHEAD;
  const int qrow0 = qtile * 16;

  const bf16* Qb = Q  + ((size_t)b_ * NHEAD + hh) * (size_t)NSEQ * DHEAD;
  const bf16* Kb = K  + ((size_t)b_ * NHEAD + hh) * (size_t)NSEQ * DHEAD;
  const bf16* Vb = Vt + ((size_t)b_ * NHEAD + hh) * (size_t)DHEAD * NSEQ;

  // Q fragments (K-dim d=64 -> two 16x32 A fragments), loaded once.
  const bf16* qrow = Qb + (size_t)(qrow0 + r) * DHEAD;
  const v16bf aq0 = load_a_frag(qrow + half * 16,      qrow + 8 + half * 16);
  const v16bf aq1 = load_a_frag(qrow + 32 + half * 16, qrow + 40 + half * 16);

  v8f o0 = {}, o1 = {}, o2 = {}, o3 = {};
  float mrow = -INFINITY, lrow = 0.f;           // row state, lanes 0..15

  float (*Sw)[32] = sS[widx];
  bf16  (*Pw)[32] = sP[widx];

  for (int m0 = 0; m0 < NSEQ; m0 += 32) {
    // ---- S = Q K^T for two 16-wide key tiles --------------------------------
    v8f s0 = {}, s1 = {};
    {
      const bf16* krow = Kb + (size_t)(m0 + r) * DHEAD;
      v16bf b0 = *(const v16bf*)(krow + half * 16);        // dd 0..31
      v16bf b1 = *(const v16bf*)(krow + 32 + half * 16);   // dd 32..63
      s0 = wmma_bf16(aq0, b0, s0);
      s0 = wmma_bf16(aq1, b1, s0);
    }
    {
      const bf16* krow = Kb + (size_t)(m0 + 16 + r) * DHEAD;
      v16bf b0 = *(const v16bf*)(krow + half * 16);
      v16bf b1 = *(const v16bf*)(krow + 32 + half * 16);
      s1 = wmma_bf16(aq0, b0, s1);
      s1 = wmma_bf16(aq1, b1, s1);
    }
    // prefetch next iteration's key tile (global_prefetch_b8)
    if (m0 + 32 < NSEQ)
      __builtin_prefetch(Kb + (size_t)(m0 + 32 + r) * DHEAD, 0, 0);

    // ---- spill S tile to per-wave LDS slice --------------------------------
#pragma unroll
    for (int rr = 0; rr < 8; ++rr) {
      Sw[rr + half * 8][r]      = s0[rr];
      Sw[rr + half * 8][16 + r] = s1[rr];
    }
    __syncthreads();

    // ---- online softmax, one row per lane (lanes 0..15) --------------------
    if (lane < 16) {
      float mx = mrow;
#pragma unroll
      for (int j = 0; j < 32; ++j) mx = fmaxf(mx, Sw[lane][j]);
      const float al = __expf(mrow - mx);
      float s = 0.f;
#pragma unroll
      for (int j = 0; j < 32; ++j) {
        float p = __expf(Sw[lane][j] - mx);
        s += p;
        Pw[lane][j] = (bf16)p;
      }
      lrow = lrow * al + s;
      mrow = mx;
      sAl[widx][lane] = al;
    }
    __syncthreads();

    // ---- rescale accumulators, O += P * V ----------------------------------
#pragma unroll
    for (int rr = 0; rr < 8; ++rr) {
      const float al = sAl[widx][rr + half * 8];
      o0[rr] *= al; o1[rr] *= al; o2[rr] *= al; o3[rr] *= al;
    }
    const bf16* prow = &Pw[r][0];
    v16bf ap = load_a_frag(prow + half * 16, prow + half * 16 + 8);

    {
      const bf16* v0 = Vb + (size_t)(0 * 16 + r) * NSEQ + m0 + half * 16;
      const bf16* v1 = Vb + (size_t)(1 * 16 + r) * NSEQ + m0 + half * 16;
      const bf16* v2 = Vb + (size_t)(2 * 16 + r) * NSEQ + m0 + half * 16;
      const bf16* v3 = Vb + (size_t)(3 * 16 + r) * NSEQ + m0 + half * 16;
      o0 = wmma_bf16(ap, *(const v16bf*)v0, o0);
      o1 = wmma_bf16(ap, *(const v16bf*)v1, o1);
      o2 = wmma_bf16(ap, *(const v16bf*)v2, o2);
      o3 = wmma_bf16(ap, *(const v16bf*)v3, o3);
      if (m0 + 32 < NSEQ) __builtin_prefetch(v0 + 32, 0, 0);
    }
  }

  // ---- epilogue: O /= l, store to attn [token][h*64+dd] bf16 ---------------
  if (lane < 16) sAl[widx][lane] = 1.0f / lrow;
  __syncthreads();

#pragma unroll
  for (int rr = 0; rr < 8; ++rr) {
    const float rinv = sAl[widx][rr + half * 8];
    const size_t row = (size_t)b_ * NSEQ + qrow0 + rr + half * 8;
    bf16* dst = attn + row * CDIM + hh * DHEAD + r;
    dst[0 * 16] = (bf16)(o0[rr] * rinv);
    dst[1 * 16] = (bf16)(o1[rr] * rinv);
    dst[2 * 16] = (bf16)(o2[rr] * rinv);
    dst[3 * 16] = (bf16)(o3[rr] * rinv);
  }
}

// ---------------------------------------------------------------------------
// Kernel 3: output projection. [8192 x 768] (bf16) x [768 x 768]^T + b -> f32
// One wave per 32x32 output block (2x2 WMMA tiles).
// ---------------------------------------------------------------------------
__global__ void __launch_bounds__(256)
proj_gemm(const bf16* __restrict__ attn, const bf16* __restrict__ wproj,
          const float* __restrict__ proj_b, float* __restrict__ out) {
  const int wave  = blockIdx.x * 8 + (threadIdx.x >> 5);
  const int lane  = threadIdx.x & 31;
  const int r     = lane & 15;
  const int half  = lane >> 4;
  const int tileC = wave % (CDIM / 32);          // 0..23
  const int tileR = wave / (CDIM / 32);          // 0..255
  const int row0  = tileR * 32;
  const int o0    = tileC * 32;

  const bf16* arow0 = attn + (size_t)(row0 + r) * CDIM;
  const bf16* arow1 = arow0 + (size_t)16 * CDIM;
  const bf16* brow0 = wproj + (size_t)(o0 + r) * CDIM;
  const bf16* brow1 = brow0 + (size_t)16 * CDIM;

  v8f acc00 = {}, acc01 = {}, acc10 = {}, acc11 = {};
#pragma unroll 2
  for (int c0 = 0; c0 < CDIM; c0 += 32) {
    v16bf a0 = load_a_frag(arow0 + c0 + half * 16, arow0 + c0 + half * 16 + 8);
    v16bf a1 = load_a_frag(arow1 + c0 + half * 16, arow1 + c0 + half * 16 + 8);
    v16bf b0 = *(const v16bf*)(brow0 + c0 + half * 16);
    v16bf b1 = *(const v16bf*)(brow1 + c0 + half * 16);
    acc00 = wmma_bf16(a0, b0, acc00);
    acc01 = wmma_bf16(a0, b1, acc01);
    acc10 = wmma_bf16(a1, b0, acc10);
    acc11 = wmma_bf16(a1, b1, acc11);
  }

  const float bias0 = proj_b[o0 + r];
  const float bias1 = proj_b[o0 + 16 + r];
#pragma unroll
  for (int rr = 0; rr < 8; ++rr) {
    const int row_a = row0 + rr + half * 8;
    const int row_b = row_a + 16;
    out[(size_t)row_a * CDIM + o0 + r]      = acc00[rr] + bias0;
    out[(size_t)row_a * CDIM + o0 + 16 + r] = acc01[rr] + bias1;
    out[(size_t)row_b * CDIM + o0 + r]      = acc10[rr] + bias0;
    out[(size_t)row_b * CDIM + o0 + 16 + r] = acc11[rr] + bias1;
  }
}

// ---------------------------------------------------------------------------
// Host launcher
// ---------------------------------------------------------------------------
extern "C" void kernel_launch(void* const* d_in, const int* in_sizes, int n_in,
                              void* d_out, int out_size, void* d_ws, size_t ws_size,
                              hipStream_t stream) {
  const float* x      = (const float*)d_in[0];   // [8,32,32,768]
  const float* qkv_w  = (const float*)d_in[1];   // [2304,768]
  const float* qkv_b  = (const float*)d_in[2];   // [2304]
  const float* proj_w = (const float*)d_in[3];   // [768,768]
  const float* proj_b = (const float*)d_in[4];   // [768]
  float* out = (float*)d_out;

  auto align_up = [](size_t v) { return (v + 255) & ~(size_t)255; };
  char* p = (char*)d_ws;
  bf16* xb    = (bf16*)p; p += align_up((size_t)ROWS * CDIM * sizeof(bf16));
  bf16* wqkv  = (bf16*)p; p += align_up((size_t)OC3  * CDIM * sizeof(bf16));
  bf16* wproj = (bf16*)p; p += align_up((size_t)CDIM * CDIM * sizeof(bf16));
  bf16* Qd    = (bf16*)p; p += align_up((size_t)ROWS * CDIM * sizeof(bf16));
  bf16* Kd    = (bf16*)p; p += align_up((size_t)ROWS * CDIM * sizeof(bf16));
  bf16* Vtd   = (bf16*)p; p += align_up((size_t)ROWS * CDIM * sizeof(bf16));
  bf16* attn  = (bf16*)p; p += align_up((size_t)ROWS * CDIM * sizeof(bf16));
  (void)ws_size; (void)n_in; (void)in_sizes; (void)out_size;

  // 0) f32 -> bf16 conversions
  {
    int n0 = ROWS * CDIM;   // 6291456
    int n1 = OC3 * CDIM;    // 1769472
    int n2 = CDIM * CDIM;   // 589824
    cvt_f32_bf16<<<n0 / 4 / 256, 256, 0, stream>>>(x, xb, n0);
    cvt_f32_bf16<<<n1 / 4 / 256, 256, 0, stream>>>(qkv_w, wqkv, n1);
    cvt_f32_bf16<<<n2 / 4 / 256, 256, 0, stream>>>(proj_w, wproj, n2);
  }
  // 1) fused QKV projection: (8192/32)*(2304/32) wave-blocks, 8 waves/block
  qkv_gemm<<<(ROWS / 32) * (OC3 / 32) / 8, 256, 0, stream>>>(
      xb, wqkv, qkv_b, Qd, Kd, Vtd);
  // 2) flash attention: 8*12*64 wave-tiles, 8 waves/block
  flash_attn<<<BATCH * NHEAD * (NSEQ / 16) / 8, 256, 0, stream>>>(
      Qd, Kd, Vtd, attn);
  // 3) output projection: (8192/32)*(768/32) wave-blocks, 8 waves/block
  proj_gemm<<<(ROWS / 32) * (CDIM / 32) / 8, 256, 0, stream>>>(
      attn, wproj, proj_b, out);
}
